// ControlTransformerBlock_28260884808307
// MI455X (gfx1250) — compile-verified, matrix operand order checked by reading, not run on hardware
//
#include <hip/hip_runtime.h>
#include <hip/hip_bf16.h>
#include <math.h>

#define B_   2
#define SI_  2048
#define ST_  512
#define S_   2560
#define D_   1536
#define H_   24
#define HD_  64
#define MLP_ 6144

typedef __attribute__((ext_vector_type(16))) __bf16 v16bf;
typedef __attribute__((ext_vector_type(8)))  float  v8f;
typedef __attribute__((ext_vector_type(4)))  unsigned int u32x4;
typedef __attribute__((ext_vector_type(8)))  int          i32x8;
typedef __attribute__((ext_vector_type(4)))  int          i32x4;

#if defined(__has_builtin)
#  if __has_builtin(__builtin_amdgcn_tensor_load_to_lds)
#    define HAVE_TDM 1
#  endif
#endif
#ifndef HAVE_TDM
#  define HAVE_TDM 0
#endif

union FragBF {
  v16bf v;
  uint4 q[2];
  unsigned short s[16];
};

__device__ __forceinline__ unsigned short f2bf(float f) {
  union { float f; unsigned int u; } c; c.f = f;
  unsigned int u = c.u;
  return (unsigned short)((u + 0x7FFFu + ((u >> 16) & 1u)) >> 16);  // RNE
}
__device__ __forceinline__ float bf2f(unsigned short s) {
  union { unsigned int u; float f; } c; c.u = ((unsigned int)s) << 16;
  return c.f;
}

// ---------------------------------------------------------------------------
// Generic bf16-WMMA GEMM:  C[z] = epilogue(A[z] @ W + bias)
// A: [gridDim.y*128, K] per batch (row-major, lda=K), W: [K][N], C: ldc=N.
// EPI 0: bias; EPI 1: bias + tanh-GELU; EPI 2: resid + gate[col]*(acc+bias)
// OBF 1: store bf16 (C is reinterpreted as ushort buffer)
// ---------------------------------------------------------------------------
template<int EPI, int OBF = 0>
__global__ __launch_bounds__(256) void gemm_wmma(
    const float* __restrict__ A, long aB,
    const float* __restrict__ W,
    const float* __restrict__ bias,
    const float* resid, long rB,
    const float* __restrict__ gate, long gB,
    float* C, long cB,
    int N, int K)
{
  __shared__ __align__(16) unsigned short sA[128][40];   // [m][k] bf16, padded
  __shared__ __align__(16) unsigned short sBT[64][40];   // [n][k] bf16 (W tile transposed)

  const int tid  = threadIdx.x;
  const int lane = tid & 31;
  const int wid  = tid >> 5;
  const int wm   = wid & 3;        // wave row group (4 x 32 rows)
  const int wn   = wid >> 2;       // wave col group (2 x 32 cols)
  const int lr   = lane & 15;
  const int hl   = lane >> 4;

  const long z  = blockIdx.z;
  const long m0 = (long)blockIdx.y * 128;
  const int  n0 = blockIdx.x * 64;

  const float* Ab = A + z * aB;

  v8f acc[2][2] = {};

  for (int k0 = 0; k0 < K; k0 += 32) {
    // --- stage A tile 128x32 (fp32 -> bf16) ---
    #pragma unroll
    for (int i = 0; i < 4; ++i) {
      int idx = tid + i * 256;
      int r   = idx >> 3;
      int c4  = (idx & 7) << 2;
      const float4 x = *(const float4*)(Ab + (m0 + r) * (long)K + k0 + c4);
      sA[r][c4+0] = f2bf(x.x); sA[r][c4+1] = f2bf(x.y);
      sA[r][c4+2] = f2bf(x.z); sA[r][c4+3] = f2bf(x.w);
    }
    // --- stage W tile 32x64, transposed into [n][k] ---
    #pragma unroll
    for (int i = 0; i < 2; ++i) {
      int idx = tid + i * 256;
      int kk  = idx >> 4;
      int c4  = (idx & 15) << 2;
      const float4 x = *(const float4*)(W + (long)(k0 + kk) * N + n0 + c4);
      sBT[c4+0][kk] = f2bf(x.x); sBT[c4+1][kk] = f2bf(x.y);
      sBT[c4+2][kk] = f2bf(x.z); sBT[c4+3][kk] = f2bf(x.w);
    }
    __syncthreads();

    FragBF af[2], bf2[2];
    #pragma unroll
    for (int t = 0; t < 2; ++t) {
      const unsigned short* p = &sA[wm*32 + t*16 + lr][hl*8];
      af[t].q[0] = *(const uint4*)p;
      af[t].q[1] = *(const uint4*)(p + 16);
    }
    #pragma unroll
    for (int u = 0; u < 2; ++u) {
      const unsigned short* p = &sBT[wn*32 + u*16 + lr][hl*8];
      bf2[u].q[0] = *(const uint4*)p;
      bf2[u].q[1] = *(const uint4*)(p + 16);
    }
    #pragma unroll
    for (int t = 0; t < 2; ++t)
      #pragma unroll
      for (int u = 0; u < 2; ++u)
        acc[t][u] = __builtin_amdgcn_wmma_f32_16x16x32_bf16(
            false, af[t].v, false, bf2[u].v, (short)0, acc[t][u], false, false);
    __syncthreads();
  }

  float*          Cb   = C + z * cB;
  unsigned short* Cb16 = (unsigned short*)C + z * cB;
  #pragma unroll
  for (int t = 0; t < 2; ++t) {
    #pragma unroll
    for (int u = 0; u < 2; ++u) {
      const int   c  = n0 + wn*32 + u*16 + lr;
      const float bv = bias[c];
      #pragma unroll
      for (int vv = 0; vv < 8; ++vv) {
        const long r = m0 + wm*32 + t*16 + hl*8 + vv;
        float x = acc[t][u][vv] + bv;
        if (EPI == 1) {
          const float x3 = x * x * x;
          x = 0.5f * x * (1.0f + tanhf(0.7978845608028654f * (x + 0.044715f * x3)));
        } else if (EPI == 2) {
          x = resid[z*rB + r*(long)N + c] + gate[z*gB + c] * x;
        }
        if (OBF) Cb16[r*(long)N + c] = f2bf(x);
        else     Cb  [r*(long)N + c] = x;
      }
    }
  }
}

// ---------------------------------------------------------------------------
// Flash attention over bf16 Q/K/V: grid = (S/128, B*H), 8 waves x 16 q-rows.
// K tiles staged global->LDS by the Tensor Data Mover when available.
// ---------------------------------------------------------------------------
__global__ __launch_bounds__(256) void flash_attn(
    const unsigned short* __restrict__ Q, const unsigned short* __restrict__ K,
    const unsigned short* __restrict__ V, float* __restrict__ O)
{
  __shared__ __align__(16) unsigned short sK [32][64];     // [key][hd] (TDM-friendly, no pad)
  __shared__ __align__(16) unsigned short sVT[64][40];     // [hd][key]
  __shared__ __align__(16) unsigned short sP [8][16][40];  // per-wave P tile [m][key]

  const int tid  = threadIdx.x;
  const int lane = tid & 31;
  const int wid  = tid >> 5;
  const int lr   = lane & 15;
  const int hl   = lane >> 4;

  const int  bh = blockIdx.y;
  const int  b  = bh / H_;
  const int  h  = bh % H_;
  const long base = (long)b * S_ * D_ + h * HD_;
  const unsigned short* Qb = Q + base;
  const unsigned short* Kb = K + base;
  const unsigned short* Vb = V + base;
  float*                Ob = O + (long)b * S_ * D_ + h * HD_;

  const int   q0    = blockIdx.x * 128 + wid * 16;
  const float scale = 0.125f;  // 1/sqrt(64), applied to scores post-WMMA

  // Q fragments: raw b128 loads, zero conversion (already bf16)
  FragBF qf[2];
  #pragma unroll
  for (int f = 0; f < 2; ++f) {
    const unsigned short* qr = Qb + (long)(q0 + lr) * D_ + f * 32;
    qf[f].q[0] = *(const uint4*)(qr + hl*8);
    qf[f].q[1] = *(const uint4*)(qr + 16 + hl*8);
  }

#if HAVE_TDM
  const unsigned ldsK_off = (unsigned)(unsigned long long)(void*)&sK[0][0];
#endif

  float m_i[8], l_i[8];
  #pragma unroll
  for (int i = 0; i < 8; ++i) { m_i[i] = -INFINITY; l_i[i] = 0.0f; }
  v8f oacc[4] = {};

  for (int kv0 = 0; kv0 < S_; kv0 += 32) {
#if HAVE_TDM
    // ---- K tile 32x64 bf16 via Tensor Data Mover (wave 0 issues, all sync) ----
    if (wid == 0) {
      const unsigned long long ga =
          (unsigned long long)(const void*)(Kb + (long)kv0 * D_);
      u32x4 g0;
      g0[0] = 1u;                                              // count=1, no gather
      g0[1] = ldsK_off;                                        // lds_addr
      g0[2] = (unsigned)ga;                                    // global_addr[31:0]
      g0[3] = (unsigned)((ga >> 32) & 0x01FFFFFFu) | (2u << 30); // addr[56:32] | type=2
      i32x8 g1;
      g1[0] = 0x00010000;        // workgroup_mask=0, data_size=1 (2 bytes)
      g1[1] = (int)(64u << 16);  // tensor_dim0 = 64
      g1[2] = (int)(32u << 16);  // tensor_dim1 = 32
      g1[3] = (int)(64u << 16);  // tile_dim0 = 64
      g1[4] = 32;                // tile_dim1 = 32, tile_dim2 = 0
      g1[5] = D_;                // tensor_dim0_stride = 1536 (elements)
      g1[6] = 0;
      g1[7] = 0;
      i32x4 z4 = {0, 0, 0, 0};
#if __clang_major__ >= 23
      i32x8 z8 = {0, 0, 0, 0, 0, 0, 0, 0};
      __builtin_amdgcn_tensor_load_to_lds(g0, g1, z4, z4, z8, 0);
#else
      __builtin_amdgcn_tensor_load_to_lds(g0, g1, z4, z4, 0);
#endif
      __builtin_amdgcn_s_wait_tensorcnt(0);
    }
#else
    // ---- fallback: direct b128 copy (bf16, no conversion) ----
    {
      const int key = tid >> 3;
      const int seg = (tid & 7) << 3;   // 8 bf16 per segment
      *(uint4*)&sK[key][seg] = *(const uint4*)(Kb + (long)(kv0 + key) * D_ + seg);
    }
#endif
    // ---- V tile scatter-transposed into [hd][key] (bf16, no conversion) ----
    {
      const int key = tid >> 3;
      const int seg = (tid & 7) << 3;
      union { uint4 q; unsigned short s[8]; } vv_;
      vv_.q = *(const uint4*)(Vb + (long)(kv0 + key) * D_ + seg);
      #pragma unroll
      for (int j = 0; j < 8; ++j) sVT[seg + j][key] = vv_.s[j];
      if (kv0 + 32 < S_) {  // prefetch next KV block
        __builtin_prefetch(Kb + (long)(kv0 + 32 + key) * D_ + seg, 0, 0);
        __builtin_prefetch(Vb + (long)(kv0 + 32 + key) * D_ + seg, 0, 0);
      }
    }
    __syncthreads();

    // S = Q K^T : two 16x16 score tiles, K-dim = head_dim (2 WMMA steps each)
    v8f sacc[2] = {};
    #pragma unroll
    for (int kk = 0; kk < 2; ++kk) {
      #pragma unroll
      for (int u = 0; u < 2; ++u) {
        FragBF kf;
        const unsigned short* p = &sK[u*16 + lr][kk*32 + hl*8];
        kf.q[0] = *(const uint4*)p;
        kf.q[1] = *(const uint4*)(p + 16);
        sacc[u] = __builtin_amdgcn_wmma_f32_16x16x32_bf16(
            false, qf[kk].v, false, kf.v, (short)0, sacc[u], false, false);
      }
    }

    // online softmax: rows in VGPR slots, columns across 16-lane halves
    #pragma unroll
    for (int vv = 0; vv < 8; ++vv) {
      float s0 = sacc[0][vv] * scale, s1 = sacc[1][vv] * scale;
      float mx = fmaxf(s0, s1);
      #pragma unroll
      for (int off = 8; off > 0; off >>= 1)
        mx = fmaxf(mx, __shfl_xor(mx, off, 32));
      const float mn = fmaxf(m_i[vv], mx);
      const float p0 = __expf(s0 - mn);
      const float p1 = __expf(s1 - mn);
      float rs = p0 + p1;
      #pragma unroll
      for (int off = 8; off > 0; off >>= 1)
        rs += __shfl_xor(rs, off, 32);
      const float corr = __expf(m_i[vv] - mn);
      m_i[vv] = mn;
      l_i[vv] = l_i[vv] * corr + rs;
      #pragma unroll
      for (int u2 = 0; u2 < 4; ++u2) oacc[u2][vv] *= corr;
      sP[wid][hl*8 + vv][lr]      = f2bf(p0);
      sP[wid][hl*8 + vv][16 + lr] = f2bf(p1);
    }
    asm volatile("" ::: "memory");  // order per-wave LDS P store -> fragment load

    // O += P V
    FragBF pf;
    {
      const unsigned short* p = &sP[wid][lr][hl*8];
      pf.q[0] = *(const uint4*)p;
      pf.q[1] = *(const uint4*)(p + 16);
    }
    #pragma unroll
    for (int u2 = 0; u2 < 4; ++u2) {
      FragBF vf;
      const unsigned short* p = &sVT[u2*16 + lr][hl*8];
      vf.q[0] = *(const uint4*)p;
      vf.q[1] = *(const uint4*)(p + 16);
      oacc[u2] = __builtin_amdgcn_wmma_f32_16x16x32_bf16(
          false, pf.v, false, vf.v, (short)0, oacc[u2], false, false);
    }
    __syncthreads();
  }

  #pragma unroll
  for (int u2 = 0; u2 < 4; ++u2)
    #pragma unroll
    for (int vv = 0; vv < 8; ++vv)
      Ob[(long)(q0 + hl*8 + vv) * D_ + u2*16 + lr] = oacc[u2][vv] / l_i[vv];
}

// ---------------------------------------------------------------------------
// Per-head RMSNorm on bf16 buffer (in-place), one warp per head
// ---------------------------------------------------------------------------
__global__ __launch_bounds__(256) void rmsnorm_heads(
    unsigned short* x, const float* __restrict__ w_img, const float* __restrict__ w_txt)
{
  const int  lane = threadIdx.x & 31;
  const int  wid  = threadIdx.x >> 5;
  const long gid  = (long)blockIdx.x * 8 + wid;
  const long b    = gid / ((long)S_ * H_);
  const long rem  = gid % ((long)S_ * H_);
  const long s    = rem / H_;
  const long h    = rem % H_;
  unsigned short* p = x + (b * S_ + s) * (long)D_ + h * HD_ + lane * 2;
  unsigned int packed = *(unsigned int*)p;
  float a = bf2f((unsigned short)(packed & 0xFFFFu));
  float c = bf2f((unsigned short)(packed >> 16));
  float ss = a * a + c * c;
  #pragma unroll
  for (int off = 16; off > 0; off >>= 1) ss += __shfl_xor(ss, off, 32);
  const float inv = rsqrtf(ss * (1.0f / 64.0f) + 1e-6f);
  const float* w = (s < SI_) ? w_img : w_txt;
  a = a * inv * w[lane * 2];
  c = c * inv * w[lane * 2 + 1];
  *(unsigned int*)p = (unsigned int)f2bf(a) | ((unsigned int)f2bf(c) << 16);
}

// ---------------------------------------------------------------------------
// SiLU + small modulation projection + elementwise modulate
// ---------------------------------------------------------------------------
__global__ void silu_kernel(const float* __restrict__ t, float* __restrict__ o, int n) {
  int i = blockIdx.x * blockDim.x + threadIdx.x;
  if (i < n) { float x = t[i]; o[i] = x / (1.0f + __expf(-x)); }
}

__global__ __launch_bounds__(256) void modproj_kernel(
    const float* __restrict__ st, const float* __restrict__ w,
    const float* __restrict__ bias, float* __restrict__ out)
{
  const int j = blockIdx.x * 256 + threadIdx.x;   // < 6*D
  const int b = blockIdx.y;
  const float* sb = st + b * D_;
  float acc = 0.0f;
  for (int kk = 0; kk < D_; ++kk)
    acc += sb[kk] * w[(long)kk * (6 * D_) + j];
  out[(long)b * 6 * D_ + j] = acc + bias[j];
}

__global__ void modulate_kernel(const float* __restrict__ x,
    const float* __restrict__ mod6, int shift_idx, int scale_idx,
    float* __restrict__ o, long SD)
{
  long i = (long)blockIdx.x * blockDim.x + threadIdx.x;
  if (i >= (long)B_ * SD) return;
  long b = i / SD;
  int  c = (int)(i % D_);
  const float* m = mod6 + b * 6 * D_;
  o[i] = x[i] * (1.0f + m[scale_idx * D_ + c]) + m[shift_idx * D_ + c];
}

// ---------------------------------------------------------------------------
extern "C" void kernel_launch(void* const* d_in, const int* in_sizes, int n_in,
                              void* d_out, int out_size, void* d_ws, size_t ws_size,
                              hipStream_t stream)
{
  const float* img_hidden = (const float*)d_in[0];
  const float* txt_hidden = (const float*)d_in[1];
  const float* temb       = (const float*)d_in[2];
  const float* img_mod_w  = (const float*)d_in[3];
  const float* img_mod_b  = (const float*)d_in[4];
  const float* txt_mod_w  = (const float*)d_in[5];
  const float* txt_mod_b  = (const float*)d_in[6];
  const float* wq  = (const float*)d_in[7];  const float* bq  = (const float*)d_in[8];
  const float* wk  = (const float*)d_in[9];  const float* bk  = (const float*)d_in[10];
  const float* wv  = (const float*)d_in[11]; const float* bv  = (const float*)d_in[12];
  const float* wo  = (const float*)d_in[13]; const float* bo  = (const float*)d_in[14];
  const float* awq = (const float*)d_in[15]; const float* abq = (const float*)d_in[16];
  const float* awk = (const float*)d_in[17]; const float* abk = (const float*)d_in[18];
  const float* awv = (const float*)d_in[19]; const float* abv = (const float*)d_in[20];
  const float* awo = (const float*)d_in[21]; const float* abo = (const float*)d_in[22];
  const float* nq_w  = (const float*)d_in[23]; const float* nk_w  = (const float*)d_in[24];
  const float* naq_w = (const float*)d_in[25]; const float* nak_w = (const float*)d_in[26];
  const float* img_w1 = (const float*)d_in[27]; const float* img_b1 = (const float*)d_in[28];
  const float* img_w2 = (const float*)d_in[29]; const float* img_b2 = (const float*)d_in[30];
  const float* txt_w1 = (const float*)d_in[31]; const float* txt_b1 = (const float*)d_in[32];
  const float* txt_w2 = (const float*)d_in[33]; const float* txt_b2 = (const float*)d_in[34];
  const float* ap_w = (const float*)d_in[35]; const float* ap_b = (const float*)d_in[36];

  // workspace layout (float-granular; qkv regions hold bf16, half-used)
  float* w  = (float*)d_ws;
  float* silu_t = w;                 w += (long)B_ * D_;
  float* im6    = w;                 w += (long)B_ * 6 * D_;
  float* tm6    = w;                 w += (long)B_ * 6 * D_;
  float* imod   = w;                 w += (long)B_ * SI_ * D_;
  float* tmod   = w;                 w += (long)B_ * ST_ * D_;
  float* qbuf   = w;                 w += (long)B_ * S_ * D_;
  float* kbuf   = w;                 w += (long)B_ * S_ * D_;
  float* vbuf   = w;                 w += (long)B_ * S_ * D_;
  float* attn   = w;                 w += (long)B_ * S_ * D_;
  float* xfi    = w;                 w += (long)B_ * SI_ * D_;
  float* xft    = w;                 w += (long)B_ * ST_ * D_;
  float* hi     = w;                 w += (long)B_ * SI_ * MLP_;
  float* ht     = w;                 w += (long)B_ * ST_ * MLP_;

  unsigned short* qb16 = (unsigned short*)qbuf;
  unsigned short* kb16 = (unsigned short*)kbuf;
  unsigned short* vb16 = (unsigned short*)vbuf;

  float* out_img = (float*)d_out;
  float* out_txt = out_img + (long)B_ * SI_ * D_;
  float* out_ctl = out_txt + (long)B_ * ST_ * D_;

  const long SDi = (long)SI_ * D_, SDt = (long)ST_ * D_, SDa = (long)S_ * D_;
  const long g6  = (long)6 * D_;

  // 1) modulation vectors
  silu_kernel<<<dim3((B_ * D_ + 255) / 256), 256, 0, stream>>>(temb, silu_t, B_ * D_);
  modproj_kernel<<<dim3(6 * D_ / 256, B_), 256, 0, stream>>>(silu_t, img_mod_w, img_mod_b, im6);
  modproj_kernel<<<dim3(6 * D_ / 256, B_), 256, 0, stream>>>(silu_t, txt_mod_w, txt_mod_b, tm6);

  // 2) pre-attention modulation (shift=idx0, scale=idx1)
  modulate_kernel<<<dim3((unsigned)((B_ * SDi + 255) / 256)), 256, 0, stream>>>(img_hidden, im6, 0, 1, imod, SDi);
  modulate_kernel<<<dim3((unsigned)((B_ * SDt + 255) / 256)), 256, 0, stream>>>(txt_hidden, tm6, 0, 1, tmod, SDt);

  // 3) QKV projections -> joint [B, S, D] bf16 buffers (img rows 0..SI, txt rows SI..S)
  dim3 gqi(D_ / 64, SI_ / 128, B_), gqt(D_ / 64, ST_ / 128, B_);
  gemm_wmma<0,1><<<gqi, 256, 0, stream>>>(imod, SDi, wq, bq, nullptr, 0, nullptr, 0, (float*)qb16, SDa, D_, D_);
  gemm_wmma<0,1><<<gqi, 256, 0, stream>>>(imod, SDi, wk, bk, nullptr, 0, nullptr, 0, (float*)kb16, SDa, D_, D_);
  gemm_wmma<0,1><<<gqi, 256, 0, stream>>>(imod, SDi, wv, bv, nullptr, 0, nullptr, 0, (float*)vb16, SDa, D_, D_);
  gemm_wmma<0,1><<<gqt, 256, 0, stream>>>(tmod, SDt, awq, abq, nullptr, 0, nullptr, 0, (float*)(qb16 + SDi), SDa, D_, D_);
  gemm_wmma<0,1><<<gqt, 256, 0, stream>>>(tmod, SDt, awk, abk, nullptr, 0, nullptr, 0, (float*)(kb16 + SDi), SDa, D_, D_);
  gemm_wmma<0,1><<<gqt, 256, 0, stream>>>(tmod, SDt, awv, abv, nullptr, 0, nullptr, 0, (float*)(vb16 + SDi), SDa, D_, D_);

  // 4) per-head RMSNorm on q, k (img vs txt weights)
  const int rms_blocks = (int)((long)B_ * S_ * H_ / 8);
  rmsnorm_heads<<<rms_blocks, 256, 0, stream>>>(qb16, nq_w, naq_w);
  rmsnorm_heads<<<rms_blocks, 256, 0, stream>>>(kb16, nk_w, nak_w);

  // 5) joint flash attention (bf16 in, f32 out)
  flash_attn<<<dim3(S_ / 128, B_ * H_), 256, 0, stream>>>(qb16, kb16, vb16, attn);

  // 6) output projections fused with residual + gate (g1 = split idx 2)
  gemm_wmma<2><<<gqi, 256, 0, stream>>>(attn, SDa, wo, bo, img_hidden, SDi, im6 + 2 * D_, g6, out_img, SDi, D_, D_);
  gemm_wmma<2><<<gqt, 256, 0, stream>>>(attn + SDi, SDa, awo, abo, txt_hidden, SDt, tm6 + 2 * D_, g6, out_txt, SDt, D_, D_);

  // 7) FFN modulation (shift=idx3, scale=idx4)
  modulate_kernel<<<dim3((unsigned)((B_ * SDi + 255) / 256)), 256, 0, stream>>>(out_img, im6, 3, 4, xfi, SDi);
  modulate_kernel<<<dim3((unsigned)((B_ * SDt + 255) / 256)), 256, 0, stream>>>(out_txt, tm6, 3, 4, xft, SDt);

  // 8) FFN: GEMM1 + GELU, GEMM2 + residual*gate (g2 = idx 5) writing final img_h/txt_h
  gemm_wmma<1><<<dim3(MLP_ / 64, SI_ / 128, B_), 256, 0, stream>>>(xfi, SDi, img_w1, img_b1, nullptr, 0, nullptr, 0, hi, (long)SI_ * MLP_, MLP_, D_);
  gemm_wmma<1><<<dim3(MLP_ / 64, ST_ / 128, B_), 256, 0, stream>>>(xft, SDt, txt_w1, txt_b1, nullptr, 0, nullptr, 0, ht, (long)ST_ * MLP_, MLP_, D_);
  gemm_wmma<2><<<gqi, 256, 0, stream>>>(hi, (long)SI_ * MLP_, img_w2, img_b2, out_img, SDi, im6 + 5 * D_, g6, out_img, SDi, D_, MLP_);
  gemm_wmma<2><<<gqt, 256, 0, stream>>>(ht, (long)ST_ * MLP_, txt_w2, txt_b2, out_txt, SDt, tm6 + 5 * D_, g6, out_txt, SDt, D_, MLP_);

  // 9) after_proj: control_out = img_h @ ap_w + ap_b
  gemm_wmma<0><<<gqi, 256, 0, stream>>>(out_img, SDi, ap_w, ap_b, nullptr, 0, nullptr, 0, out_ctl, SDi, D_, D_);

  (void)in_sizes; (void)n_in; (void)out_size; (void)ws_size;
}
